// Encoder_2310692405306
// MI455X (gfx1250) — matrix operand order repaired and checked by reading, not compiled, not run on hardware
//
#include <hip/hip_runtime.h>
#include <hip/hip_bf16.h>

typedef __attribute__((ext_vector_type(8)))  _Float16 v8h;
typedef __attribute__((ext_vector_type(16))) _Float16 v16h;
typedef __attribute__((ext_vector_type(8)))  float    v8f;

#define GAT_N   20000
#define GAT_E   320000
#define GAT_ET  (GAT_E + GAT_N)   // edges + self loops
#define GAT_H   4
#define GAT_SLOPE 0.2f

static __device__ __forceinline__ unsigned f32_sortable(float f) {
  unsigned u = __float_as_uint(f);
  return (u & 0x80000000u) ? ~u : (u | 0x80000000u);
}
static __device__ __forceinline__ float f32_unsortable(unsigned u) {
  return (u & 0x80000000u) ? __uint_as_float(u ^ 0x80000000u)
                           : __uint_as_float(~u);
}

// ---------------------------------------------------------------- fills
__global__ void fill_f32_kernel(float* __restrict__ p, long long n, float v) {
  long long i = (long long)blockIdx.x * blockDim.x + threadIdx.x;
  if (i < n) p[i] = v;
}
__global__ void fill_u32_kernel(unsigned* __restrict__ p, int n, unsigned v) {
  int i = blockIdx.x * blockDim.x + threadIdx.x;
  if (i < n) p[i] = v;
}

// ------------------------------------------------- f32 -> f16 converts
// x (N x din row-major f32) -> A (N x dinp row-major f16, zero padded)
__global__ void convert_x_kernel(const float* __restrict__ x,
                                 _Float16* __restrict__ A,
                                 int din, int dinp, long long total) {
  long long i = (long long)blockIdx.x * blockDim.x + threadIdx.x;
  if (i >= total) return;
  int k = (int)(i % dinp);
  long long n = i / dinp;
  A[i] = (k < din) ? (_Float16)x[n * din + k] : (_Float16)0.0f;
}
// W (din x kout row-major f32) -> Wt (kout x dinp row-major f16, padded)
__global__ void convert_wt_kernel(const float* __restrict__ W,
                                  _Float16* __restrict__ Wt,
                                  int din, int dinp, int kout) {
  long long total = (long long)kout * dinp;
  long long i = (long long)blockIdx.x * blockDim.x + threadIdx.x;
  if (i >= total) return;
  int k = (int)(i % dinp);
  int n = (int)(i / dinp);
  Wt[i] = (k < din) ? (_Float16)W[(long long)k * kout + n] : (_Float16)0.0f;
}

// ---------------------------------------------------------------- GEMM
// C(N x kout) = A(N x K f16, row-major) * B(K x kout), B given as
// Bt = B^T (kout x K f16, row-major).  Each wave computes a 32x64 tile
// via eight v_wmma_f32_16x16x32_f16 per K-step (A fragments reused
// across 4 B tiles -> 1.5 b128 loads per WMMA), fp32 accumulation.
static __device__ __forceinline__ v16h load_A_frag(const _Float16* __restrict__ p,
                                                   int hi) {
  // lane<16: halves = K[0..7],K[16..23]; lane>=16: K[8..15],K[24..31]
  v8h lo = *(const v8h*)(p + hi * 8);
  v8h hh = *(const v8h*)(p + 16 + hi * 8);
  v16h r;
#pragma unroll
  for (int e = 0; e < 8; ++e) { r[e] = lo[e]; r[8 + e] = hh[e]; }
  return r;
}
static __device__ __forceinline__ v16h load_B_frag(const _Float16* __restrict__ p,
                                                   int hi) {
  // lane<16: K[0..7],K[16..23] of column n; lane>=16: K[8..15],K[24..31]
  v8h lo = *(const v8h*)(p + hi * 8);
  v8h hh = *(const v8h*)(p + 16 + hi * 8);
  v16h r;
#pragma unroll
  for (int e = 0; e < 8; ++e) { r[e] = lo[e]; r[8 + e] = hh[e]; }
  return r;
}

__launch_bounds__(256)
__global__ void gat_gemm_wmma_kernel(const _Float16* __restrict__ A,
                                     const _Float16* __restrict__ Bt,
                                     float* __restrict__ C,
                                     int nrows, int K, int ncols) {
  int gwave = (int)((blockIdx.x * blockDim.x + threadIdx.x) >> 5);
  int lane  = threadIdx.x & 31;
  int colTiles = ncols >> 6;            // 64-wide column tiles
  int rowTiles = nrows >> 5;            // 32-tall row tiles
  int rt = gwave / colTiles;
  int ct = gwave - rt * colTiles;
  if (rt >= rowTiles) return;           // uniform per wave
  int lm = lane & 15;
  int hi = lane >> 4;

  const _Float16* a0 = A + (size_t)(rt * 32 + lm) * K;
  const _Float16* a1 = a0 + (size_t)16 * K;
  const _Float16* bp0 = Bt + (size_t)(ct * 64 + lm) * K;
  const _Float16* bp1 = bp0 + (size_t)16 * K;
  const _Float16* bp2 = bp0 + (size_t)32 * K;
  const _Float16* bp3 = bp0 + (size_t)48 * K;

  v8f acc0[4] = {}, acc1[4] = {};
  for (int kk = 0; kk < K; kk += 32) {
    v16h af0 = load_A_frag(a0 + kk, hi);
    v16h af1 = load_A_frag(a1 + kk, hi);
    v16h b0 = load_B_frag(bp0 + kk, hi);
    v16h b1 = load_B_frag(bp1 + kk, hi);
    v16h b2 = load_B_frag(bp2 + kk, hi);
    v16h b3 = load_B_frag(bp3 + kk, hi);
    acc0[0] = __builtin_amdgcn_wmma_f32_16x16x32_f16(false, af0, false, b0,
                                                     (short)0, acc0[0], false, false);
    acc0[1] = __builtin_amdgcn_wmma_f32_16x16x32_f16(false, af0, false, b1,
                                                     (short)0, acc0[1], false, false);
    acc0[2] = __builtin_amdgcn_wmma_f32_16x16x32_f16(false, af0, false, b2,
                                                     (short)0, acc0[2], false, false);
    acc0[3] = __builtin_amdgcn_wmma_f32_16x16x32_f16(false, af0, false, b3,
                                                     (short)0, acc0[3], false, false);
    acc1[0] = __builtin_amdgcn_wmma_f32_16x16x32_f16(false, af1, false, b0,
                                                     (short)0, acc1[0], false, false);
    acc1[1] = __builtin_amdgcn_wmma_f32_16x16x32_f16(false, af1, false, b1,
                                                     (short)0, acc1[1], false, false);
    acc1[2] = __builtin_amdgcn_wmma_f32_16x16x32_f16(false, af1, false, b2,
                                                     (short)0, acc1[2], false, false);
    acc1[3] = __builtin_amdgcn_wmma_f32_16x16x32_f16(false, af1, false, b3,
                                                     (short)0, acc1[3], false, false);
  }
  // C/D layout: VGPR r -> M = r + 8*hi, N = lane&15
  int cb = ct * 64 + lm;
  float* crow = C + (size_t)(rt * 32) * ncols;
#pragma unroll
  for (int r = 0; r < 8; ++r) {
    int m0 = r + hi * 8;
#pragma unroll
    for (int j = 0; j < 4; ++j) {
      crow[(size_t)m0 * ncols + cb + j * 16]        = acc0[j][r];
      crow[(size_t)(m0 + 16) * ncols + cb + j * 16] = acc1[j][r];
    }
  }
}

// ------------------------------------------------- attention logits
// as[n][h] = sum_c h[n][h*c+cc]*att_src[h][cc]  (same for ad)
__global__ void logits_kernel(const float* __restrict__ hbuf,
                              const float* __restrict__ att_src,
                              const float* __restrict__ att_dst,
                              float* __restrict__ as, float* __restrict__ ad,
                              int c, int kout) {
  int i = blockIdx.x * blockDim.x + threadIdx.x;
  if (i >= GAT_N * GAT_H) return;
  int n = i / GAT_H, h = i % GAT_H;
  const float* hp = hbuf + (size_t)n * kout + h * c;
  const float* sp = att_src + h * c;
  const float* dp = att_dst + h * c;
  float s = 0.f, d = 0.f;
  for (int cc = 0; cc < c; ++cc) { s += hp[cc] * sp[cc]; d += hp[cc] * dp[cc]; }
  as[i] = s;
  ad[i] = d;
}

// --------------------------------------------- edge logits + segment max
__global__ void edge_max_kernel(const int* __restrict__ src,
                                const int* __restrict__ dst,
                                const float* __restrict__ as,
                                const float* __restrict__ ad,
                                float* __restrict__ ebuf,
                                unsigned* __restrict__ mmax) {
  int j = blockIdx.x * blockDim.x + threadIdx.x;
  if (j >= GAT_ET) return;
  int s = (j < GAT_E) ? src[j] : (j - GAT_E);
  int d = (j < GAT_E) ? dst[j] : (j - GAT_E);
#pragma unroll
  for (int h = 0; h < GAT_H; ++h) {
    float e = as[s * GAT_H + h] + ad[d * GAT_H + h];
    e = (e > 0.f) ? e : GAT_SLOPE * e;
    ebuf[(size_t)j * GAT_H + h] = e;
    atomicMax(&mmax[d * GAT_H + h], f32_sortable(e));
  }
}

// --------------------------------------------- exp + segment sum
__global__ void edge_exp_kernel(const int* __restrict__ dst,
                                const float* __restrict__ ebuf,
                                const unsigned* __restrict__ mmax,
                                float* __restrict__ exbuf,
                                float* __restrict__ den) {
  int j = blockIdx.x * blockDim.x + threadIdx.x;
  if (j >= GAT_ET) return;
  int d = (j < GAT_E) ? dst[j] : (j - GAT_E);
#pragma unroll
  for (int h = 0; h < GAT_H; ++h) {
    float m = f32_unsortable(mmax[d * GAT_H + h]);
    float ex = expf(ebuf[(size_t)j * GAT_H + h] - m);
    exbuf[(size_t)j * GAT_H + h] = ex;
    atomicAdd(&den[d * GAT_H + h], ex);
  }
}

// --------------------------------------------- alpha = ex / den[dst]
__global__ void edge_alpha_kernel(const int* __restrict__ dst,
                                  const float* __restrict__ den,
                                  float* __restrict__ exbuf) {
  int i = blockIdx.x * blockDim.x + threadIdx.x;
  if (i >= GAT_ET * GAT_H) return;
  int j = i >> 2;          // GAT_H == 4
  int h = i & 3;
  int d = (j < GAT_E) ? dst[j] : (j - GAT_E);
  exbuf[i] = exbuf[i] / den[d * GAT_H + h];
}

// --------------------------------------------- weighted scatter-aggregate
__global__ void aggregate_kernel(const int* __restrict__ src,
                                 const int* __restrict__ dst,
                                 const float* __restrict__ alpha,
                                 const float* __restrict__ hbuf,
                                 float* __restrict__ agg,
                                 int koutLog2, int cLog2) {
  long long tid = (long long)blockIdx.x * blockDim.x + threadIdx.x;
  long long total = (long long)GAT_ET << koutLog2;
  if (tid >= total) return;
  int j = (int)(tid >> koutLog2);
  int t = (int)(tid & ((1 << koutLog2) - 1));
  int h = t >> cLog2;
  int s = (j < GAT_E) ? src[j] : (j - GAT_E);
  int d = (j < GAT_E) ? dst[j] : (j - GAT_E);
  float a = alpha[(size_t)j * GAT_H + h];
  float v = a * hbuf[((size_t)s << koutLog2) + t];
  atomicAdd(&agg[((size_t)d << koutLog2) + t], v);
}

// --------------------------------------------- finalize: concat layers
__global__ void finalize_concat_kernel(const float* __restrict__ agg,
                                       const float* __restrict__ bias,
                                       _Float16* __restrict__ Aout,
                                       int kout) {
  long long i = (long long)blockIdx.x * blockDim.x + threadIdx.x;
  long long total = (long long)GAT_N * kout;
  if (i >= total) return;
  float v = agg[i] + bias[i % kout];
  v = (v > 0.f) ? v : (expf(v) - 1.f);      // ELU (layers 0..2)
  Aout[i] = (_Float16)v;
}

// --------------------------------------------- finalize: mean layer (last)
__global__ void finalize_mean_kernel(const float* __restrict__ agg,
                                     const float* __restrict__ bias,
                                     float* __restrict__ out, int c) {
  int i = blockIdx.x * blockDim.x + threadIdx.x;
  if (i >= GAT_N * c) return;
  int n = i / c, cc = i % c;
  const float* ap = agg + (size_t)n * (GAT_H * c) + cc;
  float s = 0.f;
#pragma unroll
  for (int h = 0; h < GAT_H; ++h) s += ap[h * c];
  out[i] = s * (1.0f / GAT_H) + bias[cc];
}

// =====================================================================
static inline int cdiv_i(long long a, long long b) { return (int)((a + b - 1) / b); }

extern "C" void kernel_launch(void* const* d_in, const int* in_sizes, int n_in,
                              void* d_out, int out_size, void* d_ws, size_t ws_size,
                              hipStream_t stream) {
  (void)in_sizes; (void)n_in; (void)out_size; (void)ws_size;
  const float* x   = (const float*)d_in[0];
  const int*   src = (const int*)d_in[1];            // edge_index row 0
  const int*   dst = src + GAT_E;                    // edge_index row 1
  float* out = (float*)d_out;

  // workspace bump allocator (256B aligned)
  char* ws = (char*)d_ws;
  size_t off = 0;
  auto alloc = [&](size_t bytes) -> char* {
    char* p = ws + off;
    off += (bytes + 255) & ~(size_t)255;
    return p;
  };
  _Float16* A16  = (_Float16*)alloc((size_t)GAT_N * 2016 * 2);  // GEMM input
  _Float16* WT16 = (_Float16*)alloc((size_t)2016 * 512 * 2);    // B^T f16
  float*    HB   = (float*)alloc((size_t)GAT_N * 512 * 4);      // h = x@W
  float*    AGG  = (float*)alloc((size_t)GAT_N * 512 * 4);      // aggregate
  float*    AS   = (float*)alloc((size_t)GAT_N * GAT_H * 4);
  float*    AD   = (float*)alloc((size_t)GAT_N * GAT_H * 4);
  unsigned* MM   = (unsigned*)alloc((size_t)GAT_N * GAT_H * 4);
  float*    DEN  = (float*)alloc((size_t)GAT_N * GAT_H * 4);
  float*    EB   = (float*)alloc((size_t)GAT_ET * GAT_H * 4);
  float*    EX   = (float*)alloc((size_t)GAT_ET * GAT_H * 4);

  const int din_a[4]   = {2000, 512, 256, 128};
  const int dinp_a[4]  = {2016, 512, 256, 128};
  const int c_a[4]     = {128, 64, 32, 64};
  const int cLog2_a[4] = {7, 6, 5, 6};
  const int koutLog2_a[4] = {9, 8, 7, 8};
  const int T = 256;

  for (int L = 0; L < 4; ++L) {
    const int din  = din_a[L];
    const int dinp = dinp_a[L];
    const int c    = c_a[L];
    const int kout = GAT_H * c;
    const float* W    = (const float*)d_in[2 + 4 * L];
    const float* asrc = (const float*)d_in[3 + 4 * L];
    const float* adst = (const float*)d_in[4 + 4 * L];
    const float* bias = (const float*)d_in[5 + 4 * L];

    // ---- stage inputs as f16
    if (L == 0) {
      long long tot = (long long)GAT_N * dinp;
      convert_x_kernel<<<cdiv_i(tot, T), T, 0, stream>>>(x, A16, din, dinp, tot);
    }
    convert_wt_kernel<<<cdiv_i((long long)kout * dinp, T), T, 0, stream>>>(
        W, WT16, din, dinp, kout);

    // ---- GEMM: h = A16 * W  via WMMA f16->f32 (32x64 tile per wave)
    {
      int rowTiles = GAT_N / 32, colTiles = kout / 64;
      int waves = rowTiles * colTiles;
      gat_gemm_wmma_kernel<<<cdiv_i((long long)waves * 32, T), T, 0, stream>>>(
          A16, WT16, HB, GAT_N, dinp, kout);
    }

    // ---- attention logits per node/head
    logits_kernel<<<cdiv_i((long long)GAT_N * GAT_H, T), T, 0, stream>>>(
        HB, asrc, adst, AS, AD, c, kout);

    // ---- init reduction buffers
    fill_u32_kernel<<<cdiv_i((long long)GAT_N * GAT_H, T), T, 0, stream>>>(
        MM, GAT_N * GAT_H, 0x007FFFFFu);                 // sortable(-inf)
    fill_f32_kernel<<<cdiv_i((long long)GAT_N * GAT_H, T), T, 0, stream>>>(
        DEN, (long long)GAT_N * GAT_H, 0.0f);
    fill_f32_kernel<<<cdiv_i((long long)GAT_N * kout, T), T, 0, stream>>>(
        AGG, (long long)GAT_N * kout, 0.0f);

    // ---- segment softmax over incoming edges
    edge_max_kernel<<<cdiv_i(GAT_ET, T), T, 0, stream>>>(src, dst, AS, AD, EB, MM);
    edge_exp_kernel<<<cdiv_i(GAT_ET, T), T, 0, stream>>>(dst, EB, MM, EX, DEN);
    edge_alpha_kernel<<<cdiv_i((long long)GAT_ET * GAT_H, T), T, 0, stream>>>(
        dst, DEN, EX);

    // ---- weighted scatter aggregation
    {
      long long tot = (long long)GAT_ET << koutLog2_a[L];
      aggregate_kernel<<<cdiv_i(tot, T), T, 0, stream>>>(
          src, dst, EX, HB, AGG, koutLog2_a[L], cLog2_a[L]);
    }

    // ---- bias + activation / head-mean
    if (L < 3) {
      finalize_concat_kernel<<<cdiv_i((long long)GAT_N * kout, T), T, 0, stream>>>(
          AGG, bias, A16, kout);       // f16 input for next layer (ELU applied)
    } else {
      finalize_mean_kernel<<<cdiv_i((long long)GAT_N * c, T), T, 0, stream>>>(
          AGG, bias, out, c);
    }
  }
}
